// MambaTower_82016695484436
// MI455X (gfx1250) — matrix operand order repaired and checked by reading, not compiled, not run on hardware
//
#include <hip/hip_runtime.h>
#include <hip/hip_bf16.h>
#include <math.h>

// ---------------------------------------------------------------------------
// MambaTower for MI455X (gfx1250): WMMA bf16 GEMMs (16x64 tile / wave) + f32 scan
// B=4, L=1024, D=512, di=1024, n=16, dc=3, r=32, n_layers=2
// ---------------------------------------------------------------------------

typedef __attribute__((ext_vector_type(16))) __bf16 v16bf;
typedef __attribute__((ext_vector_type(8)))  float  v8f;

#define BB   4
#define LL   1024
#define DD   512
#define DI   1024
#define NN   16
#define RR   32
#define MM   (BB * LL)      // 4096 rows
#define NLAYERS 2

// ---------------------------------------------------------------------------
// WMMA GEMM:  C[M,N] = A[M,K(lda)] * W[N,K]^T  (f32 in, bf16 matmul, f32 acc)
// One wave per 16(M) x 64(N) tile: A fragment shared across 4 B fragments,
// 4 back-to-back v_wmma_f32_16x16x32_bf16 per K-step.
// A fragment (16x32 bf16): lanes 0-15 row M=l hold K 0..15,
// lanes 16-31 hold K 16..31.  B fragment (32x16): lane l = col N,
// lanes 0-15 hold K 0..15, lanes 16-31 hold K 16..31.
// epi: 0 = plain store, 1 = softplus(x + bias[col]) (dt_proj)
// ---------------------------------------------------------------------------
__global__ __launch_bounds__(32)
void gemm_wmma_bf16(const float* __restrict__ A, const float* __restrict__ W,
                    float* __restrict__ C, int M, int N, int K, int lda,
                    const float* __restrict__ bias, int epi)
{
  const int lane = threadIdx.x & 31;
  const int half = lane >> 4;      // 0: lanes 0-15, 1: lanes 16-31
  const int l    = lane & 15;
  const int tn   = blockIdx.x;     // N/64 tiles
  const int tm   = blockIdx.y;     // M/16 tiles

  const float* __restrict__ Ap = A + (size_t)(tm * 16 + l) * lda;
  const float* __restrict__ Wq[4];
#pragma unroll
  for (int q = 0; q < 4; ++q)
    Wq[q] = W + (size_t)(tn * 64 + q * 16 + l) * K + half * 16;

  v8f acc0 = {}, acc1 = {}, acc2 = {}, acc3 = {};

  for (int kb = 0; kb < K; kb += 32) {
    v16bf a;
    const int ka0 = kb + half * 16;
#pragma unroll
    for (int e = 0; e < 8; ++e)  a[e]     = (__bf16)Ap[ka0 + e];
#pragma unroll
    for (int e = 0; e < 8; ++e)  a[8 + e] = (__bf16)Ap[ka0 + 8 + e];

    v16bf b0, b1, b2, b3;
#pragma unroll
    for (int e = 0; e < 16; ++e) b0[e] = (__bf16)Wq[0][kb + e];
#pragma unroll
    for (int e = 0; e < 16; ++e) b1[e] = (__bf16)Wq[1][kb + e];
#pragma unroll
    for (int e = 0; e < 16; ++e) b2[e] = (__bf16)Wq[2][kb + e];
#pragma unroll
    for (int e = 0; e < 16; ++e) b3[e] = (__bf16)Wq[3][kb + e];

    if (kb + 32 < K) {                       // -> global_prefetch_b8
      __builtin_prefetch(Ap + kb + 32, 0, 1);
      __builtin_prefetch(Wq[0] + kb + 32, 0, 1);
      __builtin_prefetch(Wq[2] + kb + 32, 0, 1);
    }

    // 4x v_wmma_f32_16x16x32_bf16, same A operand (back-to-back matrix ops)
    acc0 = __builtin_amdgcn_wmma_f32_16x16x32_bf16(false, a, false, b0, (short)0, acc0, false, false);
    acc1 = __builtin_amdgcn_wmma_f32_16x16x32_bf16(false, a, false, b1, (short)0, acc1, false, false);
    acc2 = __builtin_amdgcn_wmma_f32_16x16x32_bf16(false, a, false, b2, (short)0, acc2, false, false);
    acc3 = __builtin_amdgcn_wmma_f32_16x16x32_bf16(false, a, false, b3, (short)0, acc3, false, false);
  }

#pragma unroll
  for (int q = 0; q < 4; ++q) {
    const v8f acc = (q == 0) ? acc0 : (q == 1) ? acc1 : (q == 2) ? acc2 : acc3;
    const int col = tn * 64 + q * 16 + l;
#pragma unroll
    for (int v = 0; v < 8; ++v) {
      const int row = tm * 16 + v + half * 8;
      float x = acc[v];
      if (epi == 1) {
        x += bias[col];
        x = (x > 20.f) ? x : log1pf(__expf(x));   // softplus
      }
      C[(size_t)row * N + col] = x;
    }
  }
}

// ---------------------------------------------------------------------------
// Embedding gather: h[m, d] = emb[x[m], d]
// ---------------------------------------------------------------------------
__global__ __launch_bounds__(256)
void embed_kernel(const int* __restrict__ x, const float* __restrict__ emb,
                  float* __restrict__ h)
{
  int i = blockIdx.x * blockDim.x + threadIdx.x;      // M*D threads
  if (i >= MM * DD) return;
  int m = i / DD, d = i % DD;
  h[i] = emb[(size_t)x[m] * DD + d];
}

// ---------------------------------------------------------------------------
// Depthwise causal conv (dc=3) + SiLU.  xi = first di cols of xz (stride 2*di)
// ---------------------------------------------------------------------------
__global__ __launch_bounds__(256)
void conv_silu_kernel(const float* __restrict__ xz,
                      const float* __restrict__ conv_w,
                      const float* __restrict__ conv_b,
                      float* __restrict__ xc)
{
  int i = blockIdx.x * blockDim.x + threadIdx.x;      // M*DI threads
  if (i >= MM * DI) return;
  int m = i / DI, d = i % DI;
  int b = m / LL, t = m % LL;
  float acc = conv_b[d];
#pragma unroll
  for (int k = 0; k < 3; ++k) {
    int ts = t + k - 2;
    if (ts >= 0)
      acc += xz[(size_t)(b * LL + ts) * (2 * DI) + d] * conv_w[d * 3 + k];
  }
  xc[i] = acc / (1.f + __expf(-acc));                 // silu
}

// ---------------------------------------------------------------------------
// Selective scan: one thread per (b, d) channel; 16 states in registers.
// y = (sum_j s_j * C_j) + x*Dp, gated by silu(z).
// ---------------------------------------------------------------------------
__global__ __launch_bounds__(256)
void scan_kernel(const float* __restrict__ xc, const float* __restrict__ dtf,
                 const float* __restrict__ dbc, const float* __restrict__ xz,
                 const float* __restrict__ A_log, const float* __restrict__ Dp,
                 float* __restrict__ ybuf)
{
  int t = blockIdx.x * blockDim.x + threadIdx.x;
  if (t >= BB * DI) return;
  int b = t / DI, d = t % DI;

  float A[NN], s[NN];
#pragma unroll
  for (int j = 0; j < NN; ++j) { A[j] = -__expf(A_log[d * NN + j]); s[j] = 0.f; }
  const float Dpd = Dp[d];

  for (int tt = 0; tt < LL; ++tt) {
    const size_t m = (size_t)b * LL + tt;
    const float dt = dtf[m * DI + d];
    const float x  = xc[m * DI + d];
    const float* __restrict__ Bm = dbc + m * 64 + RR;       // cols [32,48)
    const float* __restrict__ Cm = dbc + m * 64 + RR + NN;  // cols [48,64)
    const float dtx = dt * x;
    float y = 0.f;
#pragma unroll
    for (int j = 0; j < NN; ++j) {
      s[j] = __expf(dt * A[j]) * s[j] + dtx * Bm[j];
      y += s[j] * Cm[j];
    }
    float yv = y + x * Dpd;
    const float z = xz[m * (2 * DI) + DI + d];
    yv *= z / (1.f + __expf(-z));                           // * silu(z)
    ybuf[m * DI + d] = yv;
  }
}

// ---------------------------------------------------------------------------
// RMSNorm over D=512, one block per row
// ---------------------------------------------------------------------------
__global__ __launch_bounds__(256)
void rmsnorm_kernel(const float* __restrict__ in, const float* __restrict__ w,
                    float* __restrict__ out)
{
  __shared__ float red[256];
  const int row = blockIdx.x;
  const int tid = threadIdx.x;
  float ss = 0.f;
  for (int d = tid; d < DD; d += 256) {
    float v = in[(size_t)row * DD + d];
    ss += v * v;
  }
  red[tid] = ss; __syncthreads();
  for (int st = 128; st > 0; st >>= 1) {
    if (tid < st) red[tid] += red[tid + st];
    __syncthreads();
  }
  const float scale = rsqrtf(red[0] / (float)DD + 1e-5f);
  for (int d = tid; d < DD; d += 256)
    out[(size_t)row * DD + d] = in[(size_t)row * DD + d] * scale * w[d];
}

// ---------------------------------------------------------------------------
// Mean-pool over L, tanh, 2-wide FC.  One block per batch element.
// ---------------------------------------------------------------------------
__global__ __launch_bounds__(256)
void pool_fc_kernel(const float* __restrict__ h, const float* __restrict__ fc_w,
                    const float* __restrict__ fc_b, float* __restrict__ out)
{
  __shared__ float r0[256], r1[256];
  const int b = blockIdx.x, tid = threadIdx.x;
  float p0 = 0.f, p1 = 0.f;
  for (int d = tid; d < DD; d += 256) {
    float sum = 0.f;
    for (int t = 0; t < LL; ++t)
      sum += h[(size_t)(b * LL + t) * DD + d];
    const float tv = tanhf(sum / (float)LL);
    p0 += tv * fc_w[d];
    p1 += tv * fc_w[DD + d];
  }
  r0[tid] = p0; r1[tid] = p1; __syncthreads();
  for (int st = 128; st > 0; st >>= 1) {
    if (tid < st) { r0[tid] += r0[tid + st]; r1[tid] += r1[tid + st]; }
    __syncthreads();
  }
  if (tid == 0) {
    out[b * 2 + 0] = r0[0] + fc_b[0];
    out[b * 2 + 1] = r1[0] + fc_b[1];
  }
}

// ---------------------------------------------------------------------------
extern "C" void kernel_launch(void* const* d_in, const int* in_sizes, int n_in,
                              void* d_out, int out_size, void* d_ws, size_t ws_size,
                              hipStream_t stream) {
  (void)in_sizes; (void)n_in; (void)out_size; (void)ws_size;

  const int*   x         = (const int*)  d_in[0];
  const float* emb       = (const float*)d_in[1];
  const float* norm_w    = (const float*)d_in[2];
  const float* in_proj_w = (const float*)d_in[3];
  const float* conv_w    = (const float*)d_in[4];
  const float* conv_b    = (const float*)d_in[5];
  const float* x_proj_w  = (const float*)d_in[6];
  const float* dt_proj_w = (const float*)d_in[7];
  const float* dt_proj_b = (const float*)d_in[8];
  const float* A_log     = (const float*)d_in[9];
  const float* D_param   = (const float*)d_in[10];
  const float* out_w     = (const float*)d_in[11];
  const float* fc_w      = (const float*)d_in[12];
  const float* fc_b      = (const float*)d_in[13];
  float* out = (float*)d_out;

  // workspace carve-out (bytes)
  char* ws = (char*)d_ws;
  float* hstate = (float*)(ws + (size_t) 0        );   // M*D    f32 ( 8 MB)
  float* xzb    = (float*)(ws + (size_t) 8 * 1048576); // M*2di  (32 MB)
  float* xcb    = (float*)(ws + (size_t)40 * 1048576); // M*di   (16 MB)
  float* dbcb   = (float*)(ws + (size_t)56 * 1048576); // M*64   ( 1 MB)
  float* dtfb   = (float*)(ws + (size_t)58 * 1048576); // M*di   (16 MB)
  float* ybuf   = (float*)(ws + (size_t)74 * 1048576); // M*di   (16 MB)
  float* hnext  = (float*)(ws + (size_t)90 * 1048576); // M*D    ( 8 MB)

  // 1) embedding
  embed_kernel<<<(MM * DD + 255) / 256, 256, 0, stream>>>(x, emb, hstate);

  for (int layer = 0; layer < NLAYERS; ++layer) {
    const float* in_w_l  = in_proj_w + (size_t)layer * (2 * DI) * DD;
    const float* cw_l    = conv_w    + (size_t)layer * DI * 3;
    const float* cb_l    = conv_b    + (size_t)layer * DI;
    const float* xp_w_l  = x_proj_w  + (size_t)layer * (RR + 2 * NN) * DI;
    const float* dtw_l   = dt_proj_w + (size_t)layer * DI * RR;
    const float* dtb_l   = dt_proj_b + (size_t)layer * DI;
    const float* Alog_l  = A_log     + (size_t)layer * DI * NN;
    const float* Dp_l    = D_param   + (size_t)layer * DI;
    const float* outw_l  = out_w     + (size_t)layer * DD * DI;
    const float* nw_l    = norm_w    + (size_t)layer * DD;

    // xz = h @ in_w^T : M=4096, N=2048, K=512
    gemm_wmma_bf16<<<dim3((2 * DI) / 64, MM / 16), 32, 0, stream>>>(
        hstate, in_w_l, xzb, MM, 2 * DI, DD, DD, nullptr, 0);

    // depthwise conv + silu -> xc
    conv_silu_kernel<<<(MM * DI + 255) / 256, 256, 0, stream>>>(
        xzb, cw_l, cb_l, xcb);

    // dbc = xc @ xp_w^T : M=4096, N=64, K=1024
    gemm_wmma_bf16<<<dim3(64 / 64, MM / 16), 32, 0, stream>>>(
        xcb, xp_w_l, dbcb, MM, 64, DI, DI, nullptr, 0);

    // dt = softplus(dbc[:, :32] @ dtp_w^T + dtp_b) : M=4096, N=1024, K=32 (lda=64)
    gemm_wmma_bf16<<<dim3(DI / 64, MM / 16), 32, 0, stream>>>(
        dbcb, dtw_l, dtfb, MM, DI, RR, 64, dtb_l, 1);

    // selective scan + D skip + silu(z) gate -> ybuf
    scan_kernel<<<(BB * DI + 255) / 256, 256, 0, stream>>>(
        xcb, dtfb, dbcb, xzb, Alog_l, Dp_l, ybuf);

    // hnext = y @ out_w^T : M=4096, N=512, K=1024
    gemm_wmma_bf16<<<dim3(DD / 64, MM / 16), 32, 0, stream>>>(
        ybuf, outw_l, hnext, MM, DD, DI, DI, nullptr, 0);

    // h = rmsnorm(hnext)
    rmsnorm_kernel<<<MM, 256, 0, stream>>>(hnext, nw_l, hstate);
  }

  // mean-pool + tanh + fc
  pool_fc_kernel<<<BB, 256, 0, stream>>>(hstate, fc_w, fc_b, out);
}
